// Encoder_35424890257895
// MI455X (gfx1250) — matrix operand order repaired and checked by reading, not compiled, not run on hardware
//
#include <hip/hip_runtime.h>
#include <hip/hip_bf16.h>
#include <cstdint>

// ---------------------------------------------------------------------------
// DGCNN-style encoder for MI455X (gfx1250, wave32, WMMA).
// All GEMMs use v_wmma_f32_16x16x32_bf16 with LDS-staged 64x64 tiles laid out
// for contiguous-k fragments (k = 16*half + s); K tiled by 64 => 8 WMMA per barrier.
// A-tile staging uses the CDNA5 async global->LDS path (ASYNCcnt).
// Top-k / BN / GN / pooling are fused bandwidth-bound kernels.
// ---------------------------------------------------------------------------

typedef __bf16 bf16_t;
typedef __attribute__((ext_vector_type(8)))  __bf16 v8bf;
typedef __attribute__((ext_vector_type(16))) __bf16 v16bf;
typedef __attribute__((ext_vector_type(8)))  float  v8f;

#define NPTS 2048
#define NB   8
#define KNN  20
#define CAPC 512   // channel capacity of the concatenated feature buffer X

static inline unsigned cdivl(long a, long b) { return (unsigned)((a + b - 1) / b); }

// ---------------------------------------------------------------------------
// f32 -> bf16 conversion of a (B,C,N) feature tensor; optionally also writes
// the per-batch transposed (N,C) layout (A-operand for the kNN Gram GEMM).
// ---------------------------------------------------------------------------
__global__ void cvt_feat_bf16(const float* __restrict__ src, long srcBstride, int C,
                              bf16_t* __restrict__ dcn, bf16_t* __restrict__ dnc,
                              long total) {
  long t = (long)blockIdx.x * blockDim.x + threadIdx.x;
  if (t >= total) return;
  int n = (int)(t % NPTS);
  long r = t / NPTS;
  int c = (int)(r % C);
  int b = (int)(r / C);
  float v = src[(long)b * srcBstride + (long)c * NPTS + n];
  dcn[((long)b * C + c) * NPTS + n] = (bf16_t)v;
  if (dnc) dnc[((long)b * NPTS + n) * C + c] = (bf16_t)v;
}

__global__ void cvt_w_bf16(const float* __restrict__ src, bf16_t* __restrict__ dst, long total) {
  long t = (long)blockIdx.x * blockDim.x + threadIdx.x;
  if (t >= total) return;
  dst[t] = (bf16_t)src[t];
}

// ---------------------------------------------------------------------------
// Per-point squared norms xx[b][n] = sum_c x[b][c][n]^2
// ---------------------------------------------------------------------------
__global__ void compute_xx(const float* __restrict__ src, long srcBstride, int C,
                           float* __restrict__ xx) {
  int t = blockIdx.x * blockDim.x + threadIdx.x;
  if (t >= NB * NPTS) return;
  int n = t % NPTS, b = t / NPTS;
  const float* x = src + (long)b * srcBstride;
  float s = 0.f;
  for (int c = 0; c < C; ++c) { float v = x[(long)c * NPTS + n]; s += v * v; }
  xx[t] = s;
}

// ---------------------------------------------------------------------------
// Batched bf16 WMMA GEMM:  C[M x Ncols] = A[M x K] * B[K x Ncols] (+ bias[M])
// A row-major (lda), B row-major (ldb), C row-major f32 (ldc).
// 128 threads = 4 waves; block tile 64x64; each wave owns a 32x32 quadrant
// (2x2 of 16x16 WMMA accumulators). K tiled by 64 (8 WMMA per barrier pair).
// Fast path (K%64==0 && lda%8==0 && ldb%8==0): A tile staged with
// global_load_async_to_lds_b128 (ASYNCcnt), B tile with b128 loads +
// register transpose. Slow path: clamped zero-padded scalar staging (only
// used by the tiny block-1 GEMMs). Requires M%64==0 and Ncols%64==0.
// ---------------------------------------------------------------------------
__global__ __launch_bounds__(128)
void gemm_bf16_wmma(const bf16_t* __restrict__ A, long strideAb, int lda,
                    const bf16_t* __restrict__ B, long strideBb, int ldb,
                    float* __restrict__ C, long strideCb, int ldc,
                    int M, int Ncols, int Kdim,
                    const float* __restrict__ bias) {
  __shared__ __align__(64) bf16_t sA[64][64];  // m-major
  __shared__ __align__(64) bf16_t sB[64][64];  // n-major (transposed on write)

  int bz = blockIdx.z;
  A += (long)bz * strideAb;
  B += (long)bz * strideBb;
  C += (long)bz * strideCb;

  const int tm = blockIdx.y * 64;
  const int tn = blockIdx.x * 64;
  const int tid = threadIdx.x;
  const int wave = tid >> 5, lane = tid & 31;
  const int wm = (wave >> 1) * 32;   // wave quadrant row
  const int wn = (wave & 1) * 32;    // wave quadrant col
  const int l16 = lane & 15;
  const int lhalf = lane >> 4;
  const int kA = lhalf ? 16 : 0;     // A-fragment K start (contiguous, matches B)
  const int kB = lhalf ? 16 : 0;     // B-fragment K start (contiguous, matches A)

  const bool fast = ((Kdim & 63) == 0) && ((lda & 7) == 0) && ((ldb & 7) == 0);

  v8f acc[2][2] = {};

  for (int k0 = 0; k0 < Kdim; k0 += 64) {
    if (fast) {
      // A tile: 64x64 bf16 = 512 x 16B chunks. Each lane issues an async
      // 16B global->LDS copy (tracked by ASYNCcnt, no VGPR bounce).
#pragma unroll
      for (int it = 0; it < 4; ++it) {
        int id = tid + it * 128;           // 0..511
        int m = id >> 3, kc = (id & 7) * 8;
        unsigned ldsoff = (unsigned)(uintptr_t)&sA[m][kc];  // low 32b = LDS offset
        const bf16_t* gp = &A[(long)(tm + m) * lda + k0 + kc];
        asm volatile("global_load_async_to_lds_b128 %0, %1, off"
                     :: "v"(ldsoff), "v"(gp) : "memory");
      }
      // B tile: read 16B row segments (8 cols at one k), store transposed.
#pragma unroll
      for (int it = 0; it < 4; ++it) {
        int id = tid + it * 128;
        int k = id >> 3, n8 = (id & 7) * 8;
        v8bf v = *(const v8bf*)&B[(long)(k0 + k) * ldb + tn + n8];
#pragma unroll
        for (int j = 0; j < 8; ++j) sB[n8 + j][k] = v[j];
      }
      asm volatile("s_wait_asynccnt 0" ::: "memory");
    } else {
      // Zero-padded path; unconditional clamped loads avoid exec-mask churn.
      for (int i = tid; i < 64 * 64; i += 128) {
        int m = i >> 6, k = i & 63;
        int gk = k0 + k;
        int cl = gk < Kdim ? gk : (Kdim - 1);
        bf16_t va = A[(long)(tm + m) * lda + cl];
        sA[m][k] = (gk < Kdim) ? va : (bf16_t)0.f;
      }
      for (int i = tid; i < 64 * 64; i += 128) {
        int n = i >> 6, k = i & 63;
        int gk = k0 + k;
        int cl = gk < Kdim ? gk : (Kdim - 1);
        bf16_t vb = B[(long)cl * ldb + tn + n];
        sB[n][k] = (gk < Kdim) ? vb : (bf16_t)0.f;
      }
    }
    __syncthreads();

    // Prefetch next K tile into cache while WMMAs execute.
    if (k0 + 64 < Kdim) {
      __builtin_prefetch(&A[(long)(tm + (tid >> 1)) * lda + k0 + 64 + (tid & 1) * 32], 0, 1);
      __builtin_prefetch(&B[(long)(k0 + 64 + (tid >> 2)) * ldb + tn + (tid & 3) * 16], 0, 1);
    }

#pragma unroll
    for (int kk = 0; kk < 64; kk += 32) {
      v16bf afrag[2], bfrag[2];
#pragma unroll
      for (int s = 0; s < 2; ++s) {
        int m = wm + s * 16 + l16;
        union { v16bf v; v8bf h[2]; } ua;
        ua.h[0] = *(const v8bf*)&sA[m][kk + kA];        // K = kA .. kA+7
        ua.h[1] = *(const v8bf*)&sA[m][kk + kA + 8];    // K = kA+8 .. kA+15
        afrag[s] = ua.v;
        int n = wn + s * 16 + l16;
        bfrag[s] = *(const v16bf*)&sB[n][kk + kB];      // K = kB .. kB+15
      }
#pragma unroll
      for (int i2 = 0; i2 < 2; ++i2)
#pragma unroll
        for (int j2 = 0; j2 < 2; ++j2)
          acc[i2][j2] = __builtin_amdgcn_wmma_f32_16x16x32_bf16(
              false, afrag[i2], false, bfrag[j2], (short)0, acc[i2][j2], false, false);
    }
    __syncthreads();
  }

  // f32 C/D layout: VGPR r -> M = r + 8*lhalf, N = lane%16.
#pragma unroll
  for (int i2 = 0; i2 < 2; ++i2)
#pragma unroll
    for (int j2 = 0; j2 < 2; ++j2) {
      int rbase = tm + wm + i2 * 16 + lhalf * 8;
      int col = tn + wn + j2 * 16 + l16;
#pragma unroll
      for (int r = 0; r < 8; ++r) {
        int row = rbase + r;
        float v = acc[i2][j2][r];
        if (bias) v += bias[row];
        C[(long)row * ldc + col] = v;
      }
    }
}

// ---------------------------------------------------------------------------
// Top-20 of pd[n][m] = 2*inner - xx[n] - xx[m]; one block per row; row cached
// in LDS, 20 iterations of block-wide argmax with tie-break to lower index.
// ---------------------------------------------------------------------------
__global__ void topk20(const float* __restrict__ inner, const float* __restrict__ xx,
                       int* __restrict__ idxOut, int b) {
  __shared__ float row[NPTS];
  __shared__ float rv[256];
  __shared__ int   ri[256];
  int n = blockIdx.x, tid = threadIdx.x;
  const float* in = inner + (long)n * NPTS;
  float xn = xx[b * NPTS + n];
  for (int i = tid; i < NPTS; i += 256)
    row[i] = 2.f * in[i] - xn - xx[b * NPTS + i];
  __syncthreads();
  for (int t = 0; t < KNN; ++t) {
    float best = -1e38f; int bi = NPTS;
    for (int i = tid; i < NPTS; i += 256) {
      float v = row[i];
      if (v > best || (v == best && i < bi)) { best = v; bi = i; }
    }
    rv[tid] = best; ri[tid] = bi;
    __syncthreads();
    for (int s = 128; s > 0; s >>= 1) {
      if (tid < s) {
        if (rv[tid + s] > rv[tid] || (rv[tid + s] == rv[tid] && ri[tid + s] < ri[tid])) {
          rv[tid] = rv[tid + s]; ri[tid] = ri[tid + s];
        }
      }
      __syncthreads();
    }
    if (tid == 0) { idxOut[((long)b * NPTS + n) * KNN + t] = ri[0]; row[ri[0]] = -1e38f; }
    __syncthreads();
  }
}

// ---------------------------------------------------------------------------
// Build edge-feature matrix E[2C][N*20] (bf16) for one batch:
// rows 0..C-1 = feat - center, rows C..2C-1 = center.
// ---------------------------------------------------------------------------
__global__ void gather_edge(const float* __restrict__ src, long srcBstride, int Cin,
                            const int* __restrict__ idx, bf16_t* __restrict__ E, int b) {
  long total = (long)2 * Cin * NPTS * KNN;
  long t = (long)blockIdx.x * blockDim.x + threadIdx.x;
  if (t >= total) return;
  int j = (int)(t % KNN);
  long r = t / KNN;
  int n = (int)(r % NPTS);
  int c2 = (int)(r / NPTS);
  const float* x = src + (long)b * srcBstride;
  float v;
  if (c2 < Cin) {
    int nb = idx[((long)b * NPTS + n) * KNN + j];
    v = x[(long)c2 * NPTS + nb] - x[(long)c2 * NPTS + n];
  } else {
    v = x[(long)(c2 - Cin) * NPTS + n];
  }
  E[((long)c2 * NPTS + n) * KNN + j] = (bf16_t)v;
}

// ---------------------------------------------------------------------------
// BatchNorm2d statistics over (B, N, K) per channel; one block per channel.
// ---------------------------------------------------------------------------
__global__ void bn_stats(const float* __restrict__ Y, int O, float* __restrict__ stats) {
  int o = blockIdx.x, tid = threadIdx.x;
  long per = (long)NPTS * KNN;
  float s = 0.f, q = 0.f;
  for (int b = 0; b < NB; ++b) {
    const float* y = Y + ((long)b * O + o) * per;
    for (long i = tid; i < per; i += 256) { float v = y[i]; s += v; q += v * v; }
  }
  __shared__ float sh[256], shq[256];
  sh[tid] = s; shq[tid] = q;
  __syncthreads();
  for (int st = 128; st > 0; st >>= 1) {
    if (tid < st) { sh[tid] += sh[tid + st]; shq[tid] += shq[tid + st]; }
    __syncthreads();
  }
  if (tid == 0) { stats[o] = sh[0]; stats[O + o] = shq[0]; }
}

// ---------------------------------------------------------------------------
// Fused BN -> LeakyReLU -> max over the 20 neighbors; writes into the
// concatenated feature buffer X at channel base cbase.
// ---------------------------------------------------------------------------
__global__ void bn_lrelu_max(const float* __restrict__ Y, int O,
                             const float* __restrict__ stats,
                             const float* __restrict__ g, const float* __restrict__ bb,
                             float* __restrict__ Xout, int cbase) {
  long total = (long)NB * O * NPTS;
  long t = (long)blockIdx.x * blockDim.x + threadIdx.x;
  if (t >= total) return;
  int n = (int)(t % NPTS);
  long r = t / NPTS;
  int o = (int)(r % O);
  int b = (int)(r / O);
  float cnt = (float)NB * NPTS * KNN;
  float mean = stats[o] / cnt;
  float var = stats[O + o] / cnt - mean * mean;
  float inv = rsqrtf(var + 1e-5f);
  float ga = g[o], be = bb[o];
  const float* y = Y + (((long)b * O + o) * NPTS + n) * KNN;
  float m = -1e38f;
#pragma unroll
  for (int j = 0; j < KNN; ++j) {
    float v = (y[j] - mean) * inv * ga + be;
    v = v >= 0.f ? v : 0.2f * v;
    m = fmaxf(m, v);
  }
  Xout[((long)b * CAPC + cbase + o) * NPTS + n] = m;
}

// ---------------------------------------------------------------------------
// GroupNorm statistics per (batch, group); one block each.
// ---------------------------------------------------------------------------
__global__ void gn_stats(const float* __restrict__ Zin, int C, int groups,
                         float* __restrict__ gstats) {
  int grp = blockIdx.x, b = blockIdx.y, tid = threadIdx.x;
  int gs = C / groups;
  long per = (long)gs * NPTS;
  const float* base = Zin + ((long)b * C + (long)grp * gs) * NPTS;
  float s = 0.f, q = 0.f;
  for (long i = tid; i < per; i += 256) { float v = base[i]; s += v; q += v * v; }
  __shared__ float sh[256], shq[256];
  sh[tid] = s; shq[tid] = q;
  __syncthreads();
  for (int st = 128; st > 0; st >>= 1) {
    if (tid < st) { sh[tid] += sh[tid + st]; shq[tid] += shq[tid + st]; }
    __syncthreads();
  }
  if (tid == 0) {
    float mean = sh[0] / (float)per;
    float var = shq[0] / (float)per - mean * mean;
    gstats[(b * groups + grp) * 2] = mean;
    gstats[(b * groups + grp) * 2 + 1] = var;
  }
}

// GroupNorm apply + activation (act: 0 = LeakyReLU(0.2), 1 = ReLU).
__global__ void gn_act(const float* __restrict__ in, float* __restrict__ out,
                       const float* __restrict__ gstats,
                       const float* __restrict__ g, const float* __restrict__ bb,
                       int C, int groups, int act) {
  long total = (long)NB * C * NPTS;
  long t = (long)blockIdx.x * blockDim.x + threadIdx.x;
  if (t >= total) return;
  long r = t / NPTS;
  int c = (int)(r % C);
  int b = (int)(r / C);
  int grp = c / (C / groups);
  float mean = gstats[(b * groups + grp) * 2];
  float var = gstats[(b * groups + grp) * 2 + 1];
  float v = (in[t] - mean) * rsqrtf(var + 1e-5f) * g[c] + bb[c];
  v = (act == 1) ? fmaxf(v, 0.f) : (v >= 0.f ? v : 0.2f * v);
  out[t] = v;
}

// Global max + mean pooling over N for the 256-channel y5 -> glb[b][512].
__global__ void pool_maxmean(const float* __restrict__ Y5, float* __restrict__ glb) {
  int c = blockIdx.x, b = blockIdx.y, tid = threadIdx.x;
  const float* y = Y5 + ((long)b * 256 + c) * NPTS;
  float m = -1e38f, s = 0.f;
  for (int i = tid; i < NPTS; i += 256) { float v = y[i]; m = fmaxf(m, v); s += v; }
  __shared__ float sm[256], ss[256];
  sm[tid] = m; ss[tid] = s;
  __syncthreads();
  for (int st = 128; st > 0; st >>= 1) {
    if (tid < st) { sm[tid] = fmaxf(sm[tid], sm[tid + st]); ss[tid] += ss[tid + st]; }
    __syncthreads();
  }
  if (tid == 0) {
    glb[b * 512 + c] = sm[0];
    glb[b * 512 + 256 + c] = ss[0] / (float)NPTS;
  }
}

// Z[b][c][n]: c<512 -> loc (X buffer), c>=512 -> broadcast glb.
__global__ void concat_z(const float* __restrict__ X, const float* __restrict__ glb,
                         float* __restrict__ Z) {
  long total = (long)NB * 1024 * NPTS;
  long t = (long)blockIdx.x * blockDim.x + threadIdx.x;
  if (t >= total) return;
  int n = (int)(t % NPTS);
  long r = t / NPTS;
  int c = (int)(r % 1024);
  int b = (int)(r / 1024);
  float v = (c < 512) ? X[((long)b * CAPC + c) * NPTS + n] : glb[b * 512 + (c - 512)];
  Z[t] = v;
}

__global__ void copy_f32(const float* __restrict__ s, float* __restrict__ d, long n) {
  long t = (long)blockIdx.x * blockDim.x + threadIdx.x;
  if (t < n) d[t] = s[t];
}

// ---------------------------------------------------------------------------
// Host orchestration
// ---------------------------------------------------------------------------
extern "C" void kernel_launch(void* const* d_in, const int* in_sizes, int n_in,
                              void* d_out, int out_size, void* d_ws, size_t ws_size,
                              hipStream_t stream) {
  const float* xyz = (const float*)d_in[0];
  const float* W1 = (const float*)d_in[1];  const float* g1 = (const float*)d_in[2];  const float* b1 = (const float*)d_in[3];
  const float* W2 = (const float*)d_in[4];  const float* g2 = (const float*)d_in[5];  const float* b2 = (const float*)d_in[6];
  const float* W3 = (const float*)d_in[7];  const float* g3 = (const float*)d_in[8];  const float* b3 = (const float*)d_in[9];
  const float* W4 = (const float*)d_in[10]; const float* g4 = (const float*)d_in[11]; const float* b4 = (const float*)d_in[12];
  const float* W5 = (const float*)d_in[13]; const float* g5 = (const float*)d_in[14]; const float* b5 = (const float*)d_in[15];
  const float* Wm1 = (const float*)d_in[16]; const float* bm1 = (const float*)d_in[17];
  const float* gm1 = (const float*)d_in[18]; const float* hm1 = (const float*)d_in[19];
  const float* Wm2 = (const float*)d_in[20]; const float* bm2 = (const float*)d_in[21];
  const float* gm2 = (const float*)d_in[22]; const float* hm2 = (const float*)d_in[23];
  const float* Wm3 = (const float*)d_in[24]; const float* bm3 = (const float*)d_in[25];
  const float* gm3 = (const float*)d_in[26]; const float* hm3 = (const float*)d_in[27];

  // Workspace bump allocator (256B aligned).
  char* p = (char*)d_ws;
  auto alloc = [&](size_t bytes) -> void* {
    void* r = (void*)p;
    p += (bytes + 255) & ~(size_t)255;
    return r;
  };
  float*  X     = (float*)alloc((size_t)NB * CAPC * NPTS * 4);         // x1..x4 concat
  float*  Z1    = (float*)alloc((size_t)NB * 1024 * NPTS * 4);         // MLP buffer A
  float*  Zb    = (float*)alloc((size_t)NB * 512 * NPTS * 4);          // MLP buffer B
  float*  Y     = (float*)alloc((size_t)NB * 256 * NPTS * KNN * 4);    // edge-conv pre-max
  float*  inner = (float*)alloc((size_t)NPTS * NPTS * 4);              // per-batch Gram
  bf16_t* Bbf   = (bf16_t*)alloc((size_t)NB * 1024 * NPTS * 2);        // bf16 (C,N)
  bf16_t* BbfT  = (bf16_t*)alloc((size_t)NB * NPTS * 512 * 2);         // bf16 (N,C)
  bf16_t* E     = (bf16_t*)alloc((size_t)512 * NPTS * KNN * 2);        // per-batch edge feats
  bf16_t* Wbf   = (bf16_t*)alloc((size_t)1024 * 1024 * 2);             // bf16 weights
  float*  xx    = (float*)alloc((size_t)NB * NPTS * 4);
  int*    idx   = (int*)alloc((size_t)NB * NPTS * KNN * 4);
  float*  stats = (float*)alloc((size_t)1024 * 4);
  float*  gstats = (float*)alloc((size_t)NB * 32 * 2 * 4);
  float*  glb   = (float*)alloc((size_t)NB * 512 * 4);

  auto edge_block = [&](const float* src, long srcB, int Cin, int O,
                        const float* W, const float* g, const float* bb, int cbase) {
    cvt_w_bf16<<<cdivl((long)O * 2 * Cin, 256), 256, 0, stream>>>(W, Wbf, (long)O * 2 * Cin);
    long totF = (long)NB * Cin * NPTS;
    cvt_feat_bf16<<<cdivl(totF, 256), 256, 0, stream>>>(src, srcB, Cin, Bbf, BbfT, totF);
    compute_xx<<<cdivl((long)NB * NPTS, 256), 256, 0, stream>>>(src, srcB, Cin, xx);
    for (int b = 0; b < NB; ++b) {
      // Gram matrix: inner = Xt (N x C) * X (C x N), f32 accumulation.
      dim3 gi(NPTS / 64, NPTS / 64, 1);
      gemm_bf16_wmma<<<gi, 128, 0, stream>>>(
          BbfT + (long)b * NPTS * Cin, 0, Cin,
          Bbf + (long)b * Cin * NPTS, 0, NPTS,
          inner, 0, NPTS, NPTS, NPTS, Cin, nullptr);
      topk20<<<NPTS, 256, 0, stream>>>(inner, xx, idx, b);
    }
    for (int b = 0; b < NB; ++b) {
      long totE = (long)2 * Cin * NPTS * KNN;
      gather_edge<<<cdivl(totE, 256), 256, 0, stream>>>(src, srcB, Cin, idx, E, b);
      dim3 gy((NPTS * KNN) / 64, O / 64, 1);
      gemm_bf16_wmma<<<gy, 128, 0, stream>>>(
          Wbf, 0, 2 * Cin,
          E, 0, NPTS * KNN,
          Y + (long)b * O * NPTS * KNN, 0, NPTS * KNN,
          O, NPTS * KNN, 2 * Cin, nullptr);
    }
    bn_stats<<<O, 256, 0, stream>>>(Y, O, stats);
    long totM = (long)NB * O * NPTS;
    bn_lrelu_max<<<cdivl(totM, 256), 256, 0, stream>>>(Y, O, stats, g, bb, X, cbase);
  };

  // Four EdgeConv blocks (kNN recomputed in each block's input feature space).
  edge_block(xyz, (long)3 * NPTS, 3, 64, W1, g1, b1, 0);
  edge_block(X,                    (long)CAPC * NPTS, 64, 64,  W2, g2, b2, 64);
  edge_block(X + (long)64 * NPTS,  (long)CAPC * NPTS, 64, 128, W3, g3, b3, 128);
  edge_block(X + (long)128 * NPTS, (long)CAPC * NPTS, 128, 256, W4, g4, b4, 256);

  // y5 = W5 * loc ; GN ; LeakyReLU ; pool
  {
    long totL = (long)NB * 512 * NPTS;
    cvt_feat_bf16<<<cdivl(totL, 256), 256, 0, stream>>>(X, (long)CAPC * NPTS, 512, Bbf, nullptr, totL);
    cvt_w_bf16<<<cdivl(256 * 512, 256), 256, 0, stream>>>(W5, Wbf, 256 * 512);
    dim3 gg(NPTS / 64, 256 / 64, NB);
    gemm_bf16_wmma<<<gg, 128, 0, stream>>>(
        Wbf, 0, 512, Bbf, (long)512 * NPTS, NPTS,
        Zb, (long)256 * NPTS, NPTS, 256, NPTS, 512, nullptr);
    gn_stats<<<dim3(32, NB), 256, 0, stream>>>(Zb, 256, 32, gstats);
    gn_act<<<cdivl((long)NB * 256 * NPTS, 256), 256, 0, stream>>>(Zb, Zb, gstats, g5, b5, 256, 32, 0);
    pool_maxmean<<<dim3(256, NB), 256, 0, stream>>>(Zb, glb);
    concat_z<<<cdivl((long)NB * 1024 * NPTS, 256), 256, 0, stream>>>(X, glb, Z1);
  }

  // MLP1: 1024 -> 512, bias, GN, ReLU  (in Z1 -> Zb)
  {
    long t1 = (long)NB * 1024 * NPTS;
    cvt_feat_bf16<<<cdivl(t1, 256), 256, 0, stream>>>(Z1, (long)1024 * NPTS, 1024, Bbf, nullptr, t1);
    cvt_w_bf16<<<cdivl(512 * 1024, 256), 256, 0, stream>>>(Wm1, Wbf, 512 * 1024);
    dim3 gg(NPTS / 64, 512 / 64, NB);
    gemm_bf16_wmma<<<gg, 128, 0, stream>>>(
        Wbf, 0, 1024, Bbf, (long)1024 * NPTS, NPTS,
        Zb, (long)512 * NPTS, NPTS, 512, NPTS, 1024, bm1);
    gn_stats<<<dim3(32, NB), 256, 0, stream>>>(Zb, 512, 32, gstats);
    gn_act<<<cdivl((long)NB * 512 * NPTS, 256), 256, 0, stream>>>(Zb, Zb, gstats, gm1, hm1, 512, 32, 1);
  }

  // MLP2: 512 -> 256, bias, GN, ReLU  (Zb -> Z1 region)
  {
    long t2 = (long)NB * 512 * NPTS;
    cvt_feat_bf16<<<cdivl(t2, 256), 256, 0, stream>>>(Zb, (long)512 * NPTS, 512, Bbf, nullptr, t2);
    cvt_w_bf16<<<cdivl(256 * 512, 256), 256, 0, stream>>>(Wm2, Wbf, 256 * 512);
    dim3 gg(NPTS / 64, 256 / 64, NB);
    gemm_bf16_wmma<<<gg, 128, 0, stream>>>(
        Wbf, 0, 512, Bbf, (long)512 * NPTS, NPTS,
        Z1, (long)256 * NPTS, NPTS, 256, NPTS, 512, bm2);
    gn_stats<<<dim3(32, NB), 256, 0, stream>>>(Z1, 256, 32, gstats);
    gn_act<<<cdivl((long)NB * 256 * NPTS, 256), 256, 0, stream>>>(Z1, Z1, gstats, gm2, hm2, 256, 32, 1);
  }

  // MLP3: 256 -> 128, bias, GN, ReLU -> emb written straight to d_out.
  {
    long t3 = (long)NB * 256 * NPTS;
    cvt_feat_bf16<<<cdivl(t3, 256), 256, 0, stream>>>(Z1, (long)256 * NPTS, 256, Bbf, nullptr, t3);
    cvt_w_bf16<<<cdivl(128 * 256, 256), 256, 0, stream>>>(Wm3, Wbf, 128 * 256);
    dim3 gg(NPTS / 64, 128 / 64, NB);
    gemm_bf16_wmma<<<gg, 128, 0, stream>>>(
        Wbf, 0, 256, Bbf, (long)256 * NPTS, NPTS,
        Zb, (long)128 * NPTS, NPTS, 128, NPTS, 256, bm3);
    gn_stats<<<dim3(32, NB), 256, 0, stream>>>(Zb, 128, 32, gstats);
    gn_act<<<cdivl((long)NB * 128 * NPTS, 256), 256, 0, stream>>>(
        Zb, (float*)d_out, gstats, gm3, hm3, 128, 32, 1);
  }

  // glb -> tail of d_out (B x 512 x 1).
  copy_f32<<<cdivl((long)NB * 512, 256), 256, 0, stream>>>(
      glb, (float*)d_out + (long)NB * 128 * NPTS, (long)NB * 512);
}
